// PentanacciKANLayer_30331059044963
// MI455X (gfx1250) — matrix operand order repaired and checked by reading, not compiled, not run on hardware
//
#include <hip/hip_runtime.h>
#include <hip/hip_bf16.h>

typedef _Float16 v16h __attribute__((ext_vector_type(16)));
typedef _Float16 v8h  __attribute__((ext_vector_type(8)));
typedef float    v8f  __attribute__((ext_vector_type(8)));

#define B_DIM   4096
#define I_DIM   1024
#define O_DIM   1024
#define KTOT    8192          // 8 useful degrees (d=1..8), d=0 basis is identically zero
#define BK      32
#define BM      128
#define BN      128
#define LDS_ROW 40            // 32 f16 + 8 f16 pad -> 80B row stride, conflict-free b128 reads

#define STAGE_E      10240u   // f16 elements per stage (A 128*40 + B 128*40)
#define STAGE_BYTES  20480u
#define B_OFF_E      5120u    // B tile offset within a stage, in f16 elements
#define B_OFF_BYTES  10240u
#define NK           (KTOT / BK)   // 256
#define CHUNK1_GB    ((unsigned long long)64 * KTOT * 2)  // +64 rows in global (1 MiB)
#define CHUNK1_LDS   (64u * 80u)                          // +64 rows in LDS (5120 B)

// ---------------------------------------------------------------------------
// Kernel 1: xs = sigmoid(x); pentanacci basis (fp32); store A[b, (d-1)*1024+i] as f16
// ---------------------------------------------------------------------------
__global__ void build_basis_kernel(const float* __restrict__ x,
                                   _Float16* __restrict__ A) {
    int idx = blockIdx.x * blockDim.x + threadIdx.x;      // b*1024 + i
    int b = idx >> 10;
    int i = idx & 1023;
    float xv = x[idx];
    float s  = 1.0f / (1.0f + __expf(-xv));
    float t1 = 1.0f;
    float t2 = s;
    float t3 = s;
    float t4 = s * s;
    float t5 = fmaf(s, t4, t3 + t2 + t1);          // + b0 (=0)
    float t6 = fmaf(s, t5, t4 + t3 + t2 + t1);
    float t7 = fmaf(s, t6, t5 + t4 + t3 + t2);
    float t8 = fmaf(s, t7, t6 + t5 + t4 + t3);
    _Float16* row = A + (size_t)b * KTOT + i;
    row[0 * 1024] = (_Float16)t1;
    row[1 * 1024] = (_Float16)t2;
    row[2 * 1024] = (_Float16)t3;
    row[3 * 1024] = (_Float16)t4;
    row[4 * 1024] = (_Float16)t5;
    row[5 * 1024] = (_Float16)t6;
    row[6 * 1024] = (_Float16)t7;
    row[7 * 1024] = (_Float16)t8;
}

// ---------------------------------------------------------------------------
// Kernel 2: coeffs[i,o,d] (f32) -> Wt[o, (d-1)*1024+i] (f16), N-major / K-contiguous
// ---------------------------------------------------------------------------
__global__ void pack_coeffs_kernel(const float* __restrict__ coeffs,
                                   _Float16* __restrict__ Wt) {
    int idx = blockIdx.x * blockDim.x + threadIdx.x;      // o*1024 + i, i fastest
    int i = idx & 1023;
    int o = idx >> 10;
    const float* src = coeffs + ((size_t)i * O_DIM + o) * 9;
    _Float16* dst = Wt + (size_t)o * KTOT + i;
    #pragma unroll
    for (int d = 1; d <= 8; ++d) {
        dst[(d - 1) * 1024] = (_Float16)src[d];           // coalesced b16 stores over i
    }
}

// ---------------------------------------------------------------------------
// Kernel 3: Y[4096,1024] = A[4096,8192] * Wt^T   (f16 in, f32 accumulate, WMMA)
// Block: 256 threads = 8 waves; block tile 128x128; wave tile 32x64 (2x4 frags).
// Global -> LDS via GLOBAL_LOAD_ASYNC_TO_LDS_B128 (ASYNCcnt), double-buffered.
// k-loop kept at unroll 1 so accumulators stay pinned (no mov/nop seams).
// ---------------------------------------------------------------------------
__global__ __launch_bounds__(256)
void gemm_wmma_kernel(const _Float16* __restrict__ A,
                      const _Float16* __restrict__ Wt,
                      float* __restrict__ out) {
    __shared__ _Float16 lds[2 * STAGE_E];   // [stage][A 128x40 | B 128x40] f16

    const int tid  = threadIdx.x;
    const int lane = tid & 31;
    const int wave = tid >> 5;
    const int r    = lane & 15;
    const int half = lane >> 4;
    const int wm   = wave >> 1;     // 0..3 -> 4 * 32 rows
    const int wn   = wave & 1;      // 0..1 -> 2 * 64 cols

    const int m_base = blockIdx.y * BM;
    const int n_base = blockIdx.x * BN;

    // chunk mapping: thread handles b128 chunks c and c+256 of each 512-chunk tile;
    // chunk c+256 == chunk c shifted by +64 rows (constant offsets).
    const int ar0 = tid >> 2, ac0 = tid & 3;

    const unsigned lds_base = (unsigned)(unsigned long long)(void*)&lds[0];
    const unsigned aoff0 = (unsigned)(ar0 * 80 + ac0 * 16);
    const unsigned boff0 = B_OFF_BYTES + aoff0;

    const unsigned long long ga0 =
        (unsigned long long)(const char*)(A + (size_t)m_base * KTOT)
        + (unsigned long long)ar0 * (KTOT * 2) + (unsigned)(ac0 * 16);
    const unsigned long long gb0 =
        (unsigned long long)(const char*)(Wt + (size_t)n_base * KTOT)
        + (unsigned long long)ar0 * (KTOT * 2) + (unsigned)(ac0 * 16);

    #define ASYNC_B128(ldsoff, gaddr)                                         \
        asm volatile("global_load_async_to_lds_b128 %0, %1, off"              \
                     :: "v"(ldsoff), "v"(gaddr) : "memory")

    auto issue_tile = [&](unsigned stage, int kt) {
        const unsigned sb = lds_base + stage * STAGE_BYTES;
        const unsigned long long kb = (unsigned long long)kt * (BK * 2);  // 64 B / K-step
        ASYNC_B128(sb + aoff0,              ga0 + kb);
        ASYNC_B128(sb + aoff0 + CHUNK1_LDS, ga0 + kb + CHUNK1_GB);
        ASYNC_B128(sb + boff0,              gb0 + kb);
        ASYNC_B128(sb + boff0 + CHUNK1_LDS, gb0 + kb + CHUNK1_GB);
    };

    v8f acc[2][4];
    #pragma unroll
    for (int mi = 0; mi < 2; ++mi)
        #pragma unroll
        for (int ni = 0; ni < 4; ++ni)
            acc[mi][ni] = (v8f){0.f, 0.f, 0.f, 0.f, 0.f, 0.f, 0.f, 0.f};

    union Frag16 { v16h v; v8h h[2]; };

    auto compute_stage = [&](unsigned stage) {
        const _Float16* Acur = lds + stage * STAGE_E;
        const _Float16* Bcur = Acur + B_OFF_E;
        Frag16 bf[4];
        #pragma unroll
        for (int ni = 0; ni < 4; ++ni) {
            const int brow = (wn * 64 + ni * 16 + r) * LDS_ROW;
            // lanes 0-15: K 0-15 ; lanes 16-31: K 16-31 (contiguous per lane)
            bf[ni].h[0] = *(const v8h*)(&Bcur[brow + half * 16]);
            bf[ni].h[1] = *(const v8h*)(&Bcur[brow + half * 16 + 8]);
        }
        #pragma unroll
        for (int mi = 0; mi < 2; ++mi) {
            const int arow = (wm * 32 + mi * 16 + r) * LDS_ROW;
            Frag16 af;
            // lanes 0-15: K 0-15 ; lanes 16-31: K 16-31 (contiguous per lane)
            af.h[0] = *(const v8h*)(&Acur[arow + half * 16]);
            af.h[1] = *(const v8h*)(&Acur[arow + half * 16 + 8]);
            #pragma unroll
            for (int ni = 0; ni < 4; ++ni)
                acc[mi][ni] = __builtin_amdgcn_wmma_f32_16x16x32_f16(
                    false, af.v, false, bf[ni].v,
                    (short)0, acc[mi][ni], false, false);
        }
    };

    issue_tile(0u, 0);   // prologue: tile 0 -> stage 0 (ASYNCcnt = 4)

    #pragma unroll 1
    for (int kt = 0; kt < NK - 1; ++kt) {
        const unsigned cur = (unsigned)kt & 1u;
        issue_tile(cur ^ 1u, kt + 1);                     // next tile in flight
        asm volatile("s_wait_asynccnt 0x4" ::: "memory"); // tile kt's 4 loads done
        __syncthreads();                                  // publish stage `cur`
        compute_stage(cur);
        __syncthreads();                                  // stage `cur` safe to refill
    }
    // peeled final iteration: nothing left to prefetch
    asm volatile("s_wait_asynccnt 0x0" ::: "memory");
    __syncthreads();
    compute_stage((NK - 1) & 1u);

    // epilogue: C layout — VGPR v, lane<16 -> row v, col lane; lane>=16 -> row 8+v
    #pragma unroll
    for (int mi = 0; mi < 2; ++mi) {
        #pragma unroll
        for (int ni = 0; ni < 4; ++ni) {
            const int col = n_base + wn * 64 + ni * 16 + r;
            const int rw  = m_base + wm * 32 + mi * 16 + half * 8;
            #pragma unroll
            for (int v = 0; v < 8; ++v)
                out[(size_t)(rw + v) * O_DIM + col] = acc[mi][ni][v];
        }
    }
}

// ---------------------------------------------------------------------------
extern "C" void kernel_launch(void* const* d_in, const int* in_sizes, int n_in,
                              void* d_out, int out_size, void* d_ws, size_t ws_size,
                              hipStream_t stream) {
    const float* x      = (const float*)d_in[0];   // 4096*1024
    const float* coeffs = (const float*)d_in[1];   // 1024*1024*9
    float* out          = (float*)d_out;           // 4096*1024 (f32)

    _Float16* A  = (_Float16*)d_ws;                                     // 64 MB
    _Float16* Wt = (_Float16*)((char*)d_ws + (size_t)B_DIM * KTOT * 2); // 16 MB

    build_basis_kernel<<<(B_DIM * I_DIM) / 256, 256, 0, stream>>>(x, A);
    pack_coeffs_kernel<<<(O_DIM * I_DIM) / 256, 256, 0, stream>>>(coeffs, Wt);

    dim3 grid(O_DIM / BN, B_DIM / BM);   // (8, 32)
    gemm_wmma_kernel<<<grid, 256, 0, stream>>>(A, Wt, out);
}
